// NeuroFlexNet_71536975282464
// MI455X (gfx1250) — compile-verified, matrix operand order checked by reading, not run on hardware
//
#include <hip/hip_runtime.h>
#include <stdint.h>

// ---------------- problem constants ----------------
#define SEQ    512
#define BATCH  128
#define IN_F   512
#define HDIM   512
#define OUT_F  512
#define GDIM   (4 * HDIM)   // 2048 gate outputs per cell

#define KCHUNK 64           // K staged per LDS buffer
#define APAD   72           // padded LDS row stride (halves): conflict-free b128

// ---------------- WMMA types (CDNA5 / gfx1250, wave32) ----------------
typedef __attribute__((ext_vector_type(16))) __bf16 bf16x16;
typedef __attribute__((ext_vector_type(8)))  float  f32x8;

__device__ __forceinline__ unsigned short f2b(float f) {
    // round-to-nearest-even fp32 -> bf16
    unsigned u = __builtin_bit_cast(unsigned, f);
    u += 0x7FFFu + ((u >> 16) & 1u);
    return (unsigned short)(u >> 16);
}
__device__ __forceinline__ float sigmoidf(float x) {
    return 1.0f / (1.0f + __expf(-x));
}

// ---------------- prep kernels ----------------
__global__ void k_cvt_bf16(const float* __restrict__ src,
                           unsigned short* __restrict__ dst, int n) {
    for (int i = blockIdx.x * blockDim.x + threadIdx.x; i < n;
         i += gridDim.x * blockDim.x)
        dst[i] = f2b(src[i]);
}

__global__ void k_add_bias(const float* __restrict__ a,
                           const float* __restrict__ b,
                           float* __restrict__ c, int n) {
    int i = blockIdx.x * blockDim.x + threadIdx.x;
    if (i < n) c[i] = a[i] + b[i];
}

__global__ void k_zero_f32(float* __restrict__ p, int n) {
    for (int i = blockIdx.x * blockDim.x + threadIdx.x; i < n;
         i += gridDim.x * blockDim.x)
        p[i] = 0.0f;
}

// ---------------- dual-source bf16 WMMA GEMM ----------------
// C(128 x N) = A1(128 x K1) @ W1^T + A2(128 x K2) @ W2^T + bias, optional sigmoid.
// W matrices: (N x K) row-major bf16 (B-columns = W-rows, contiguous K).
// A: fp32, converted to bf16 while staged through a double-buffered LDS panel.
// Each wave owns 1 N-tile x all 8 M-tiles (8 accumulators): every weight
// fragment is loaded once per GEMM and feeds 8 back-to-back WMMAs.
// grid = (N/128), block = 256 (8 wave32s).
__global__ __launch_bounds__(256)
void k_gemm_dual(const float* __restrict__ A1, int lda1,
                 const float* __restrict__ A2, int lda2,
                 const unsigned short* __restrict__ W1, int ldw1,
                 const unsigned short* __restrict__ W2, int ldw2,
                 const float* __restrict__ bias,
                 int K1, int K2,
                 float* __restrict__ Cf, int ldc, int act)
{
    __shared__ unsigned short As[2][128 * APAD];   // 2 x 128 rows x 64 halves (padded)

    const int tid  = threadIdx.x;
    const int lane = tid & 31;
    const int wave = tid >> 5;
    const int lr   = lane & 15;                    // row/col within 16x16 tile
    const int kh   = (lane >> 4) << 4;             // K half select: 0 / 16
    const int col  = ((blockIdx.x << 3) + wave) << 4;

    // staging role: 2 threads per batch row, 32 halves (8 x float4) each
    const int srow = tid >> 1;                     // 0..127
    const int soff = (tid & 1) << 5;               // 0 / 32

    f32x8 acc[8];
    #pragma unroll
    for (int m = 0; m < 8; ++m) acc[m] = (f32x8){0.f,0.f,0.f,0.f,0.f,0.f,0.f,0.f};

    for (int phase = 0; phase < 2; ++phase) {
        const int K = phase ? K2 : K1;
        if (K == 0) continue;
        const float* __restrict__ Ap = phase ? A2 : A1;
        const unsigned short* __restrict__ Wp = phase ? W2 : W1;
        const int lda = phase ? lda2 : lda1;
        const int ldw = phase ? ldw2 : ldw1;
        const unsigned short* __restrict__ wrow = Wp + (size_t)(col + lr) * ldw;

        // stage chunk 0 into buffer 0
        {
            const float* src = Ap + (size_t)srow * lda + soff;
            unsigned short* dst = &As[0][srow * APAD + soff];
            #pragma unroll
            for (int i = 0; i < 8; ++i) {
                float4 v = *(const float4*)(src + i * 4);
                unsigned p0 = (unsigned)f2b(v.x) | ((unsigned)f2b(v.y) << 16);
                unsigned p1 = (unsigned)f2b(v.z) | ((unsigned)f2b(v.w) << 16);
                *(uint2*)(dst + i * 4) = make_uint2(p0, p1);
            }
        }

        int buf = 0;
        for (int kc = 0; kc < K; kc += KCHUNK) {
            __syncthreads();   // staged chunk visible; previous buffer free

            // overlap: stage next chunk into the other buffer while computing
            if (kc + KCHUNK < K) {
                const float* src = Ap + (size_t)srow * lda + (kc + KCHUNK) + soff;
                unsigned short* dst = &As[buf ^ 1][srow * APAD + soff];
                #pragma unroll
                for (int i = 0; i < 8; ++i) {
                    float4 v = *(const float4*)(src + i * 4);
                    unsigned p0 = (unsigned)f2b(v.x) | ((unsigned)f2b(v.y) << 16);
                    unsigned p1 = (unsigned)f2b(v.z) | ((unsigned)f2b(v.w) << 16);
                    *(uint2*)(dst + i * 4) = make_uint2(p0, p1);
                }
            }
            __builtin_prefetch(wrow + kc + 2 * KCHUNK, 0, 3);   // global_prefetch_b8

            // compute: 2 k-steps x 8 m-tiles = 16 WMMAs on this chunk
            #pragma unroll
            for (int ks = 0; ks < 2; ++ks) {
                const bf16x16 bfrag =
                    *(const bf16x16*)(wrow + kc + ks * 32 + kh);   // 1 global load
                #pragma unroll
                for (int m = 0; m < 8; ++m) {                      // reused 8x
                    const bf16x16 afrag = *(const bf16x16*)
                        (&As[buf][(m * 16 + lr) * APAD + ks * 32 + kh]);
                    acc[m] = __builtin_amdgcn_wmma_f32_16x16x32_bf16(
                        false, afrag, false, bfrag, (short)0, acc[m], false, false);
                }
            }
            buf ^= 1;
        }
        __syncthreads();   // drain before next phase reuses the buffers
    }

    // epilogue: C layout — VGPR v, lanes 0-15 -> M=v, lanes 16-31 -> M=8+v
    const int c   = col + lr;
    const float bv = bias[c];
    const int rof = (lane >> 4) << 3;
    #pragma unroll
    for (int m = 0; m < 8; ++m) {
        #pragma unroll
        for (int v = 0; v < 8; ++v) {
            float val = acc[m][v] + bv;
            if (act) val = sigmoidf(val);
            Cf[(size_t)(m * 16 + rof + v) * ldc + c] = val;
        }
    }
}

// ---------------- fused NF-cell elementwise + LayerNorm ----------------
// gates: (BATCH x 4H) fp32 [i|f|g|o].  One block per batch row, 256 thr x 2 elems.
// Updates hstate/cstate in place (they live in d_out's state slots).
__global__ __launch_bounds__(256)
void k_nf_update(const float* __restrict__ gates,
                 const float* __restrict__ xin, int ldx,
                 float* __restrict__ hstate,
                 float* __restrict__ cstate,
                 const float* __restrict__ Avec,
                 const float* __restrict__ wtau,
                 const float* __restrict__ gamma,
                 const float* __restrict__ beta)
{
    const int b = blockIdx.x;
    __shared__ float s1[8], s2[8];
    float xc[2], c2v[2];
    float lsum = 0.f, lsq = 0.f;

    #pragma unroll
    for (int u = 0; u < 2; ++u) {
        const int j = threadIdx.x + (u << 8);
        const float gi = gates[(size_t)b * GDIM + j];
        const float gf = gates[(size_t)b * GDIM + HDIM + j];
        const float gg = gates[(size_t)b * GDIM + 2 * HDIM + j];
        const float go = gates[(size_t)b * GDIM + 3 * HDIM + j];
        const float cp = cstate[(size_t)b * HDIM + j];
        const float c2 = sigmoidf(gf) * cp + sigmoidf(gi) * tanhf(gg);
        const float hl = sigmoidf(go) * tanhf(c2);
        const float x  = xin[(size_t)b * ldx + j];
        const float sx = sigmoidf(x);
        const float e  = __expf(-(wtau[j] + sx) + 1e-5f);   // DT = 1
        const float v  = (hl - Avec[j]) * e * (1.0f - sx)   // s(-x) = 1 - s(x)
                         + Avec[j] + hstate[(size_t)b * HDIM + j];
        xc[u] = v; c2v[u] = c2;
        lsum += v; lsq += v * v;
    }
    // wave32 reduction + cross-wave via LDS
    #pragma unroll
    for (int off = 16; off > 0; off >>= 1) {
        lsum += __shfl_xor(lsum, off, 32);
        lsq  += __shfl_xor(lsq,  off, 32);
    }
    if ((threadIdx.x & 31) == 0) {
        s1[threadIdx.x >> 5] = lsum;
        s2[threadIdx.x >> 5] = lsq;
    }
    __syncthreads();
    float sum = 0.f, sq = 0.f;
    #pragma unroll
    for (int w = 0; w < 8; ++w) { sum += s1[w]; sq += s2[w]; }

    const float mu   = sum * (1.0f / HDIM);
    const float var  = sq * (1.0f / HDIM) - mu * mu;   // biased, matches torch LN
    const float rstd = rsqrtf(var + 1e-5f);

    #pragma unroll
    for (int u = 0; u < 2; ++u) {
        const int j = threadIdx.x + (u << 8);
        const float o = (xc[u] - mu) * rstd * gamma[j] + beta[j];
        hstate[(size_t)b * HDIM + j] = o;
        cstate[(size_t)b * HDIM + j] = c2v[u];
    }
}

// ---------------- host launcher ----------------
extern "C" void kernel_launch(void* const* d_in, const int* in_sizes, int n_in,
                              void* d_out, int out_size, void* d_ws, size_t ws_size,
                              hipStream_t stream)
{
    (void)in_sizes; (void)n_in; (void)out_size; (void)ws_size;

    const float* inputs = (const float*)d_in[0];
    // cell params: order per _make_cell: Wih, Whh, bih, bhh, A, w_tau, gamma, beta
    // cells: 0=perc, 1=gw, 2=obs, 3=mir_act, 4=ae
    const float *Wih[5], *Whh[5], *bih[5], *bhh[5], *Av[5], *wt[5], *gm[5], *bt[5];
    for (int c = 0; c < 5; ++c) {
        int base = 1 + 8 * c;
        Wih[c] = (const float*)d_in[base + 0];
        Whh[c] = (const float*)d_in[base + 1];
        bih[c] = (const float*)d_in[base + 2];
        bhh[c] = (const float*)d_in[base + 3];
        Av[c]  = (const float*)d_in[base + 4];
        wt[c]  = (const float*)d_in[base + 5];
        gm[c]  = (const float*)d_in[base + 6];
        bt[c]  = (const float*)d_in[base + 7];
    }
    const float* gwW  = (const float*)d_in[41];
    const float* gwb  = (const float*)d_in[42];
    const float* mirW = (const float*)d_in[43];
    const float* mirb = (const float*)d_in[44];
    const float* outW = (const float*)d_in[45];
    const float* outb = (const float*)d_in[46];

    // -------- workspace layout (~25 MB) --------
    char* wsp = (char*)d_ws;
    size_t off = 0;
    auto alloc = [&](size_t bytes) -> void* {
        void* r = wsp + off;
        off += (bytes + 255) & ~(size_t)255;
        return r;
    };
    unsigned short *wihB[5], *whhB[5];
    float* biasC[5];
    for (int c = 0; c < 5; ++c) {
        wihB[c]  = (unsigned short*)alloc((size_t)GDIM * HDIM * 2);
        whhB[c]  = (unsigned short*)alloc((size_t)GDIM * HDIM * 2);
        biasC[c] = (float*)alloc((size_t)GDIM * 4);
    }
    unsigned short* gwWB  = (unsigned short*)alloc((size_t)HDIM * HDIM * 2);
    unsigned short* mirWB = (unsigned short*)alloc((size_t)HDIM * 2 * HDIM * 2);
    unsigned short* outWB = (unsigned short*)alloc((size_t)OUT_F * HDIM * 2);
    float* dec   = (float*)alloc((size_t)BATCH * HDIM * 4);
    float* mir   = (float*)alloc((size_t)BATCH * HDIM * 4);
    float* gates = (float*)alloc((size_t)BATCH * GDIM * 4);

    // -------- state slots live directly in d_out (tuple order) --------
    float* outp   = (float*)d_out;
    float* states = outp + (size_t)SEQ * BATCH * OUT_F;
    const size_t BH = (size_t)BATCH * HDIM;
    float* h_perc = states + 0 * BH;  float* c_perc = states + 1 * BH;
    float* h_obs  = states + 2 * BH;  float* c_obs  = states + 3 * BH;
    float* h_act  = states + 4 * BH;  float* c_act  = states + 5 * BH;
    float* h_gw   = states + 6 * BH;  float* c_gw   = states + 7 * BH;
    float* h_ae   = states + 8 * BH;  float* c_ae   = states + 9 * BH;

    // -------- prep: weights -> bf16 (L2-resident for the whole sequence) --------
    for (int c = 0; c < 5; ++c) {
        k_cvt_bf16<<<512, 256, 0, stream>>>(Wih[c], wihB[c], GDIM * HDIM);
        k_cvt_bf16<<<512, 256, 0, stream>>>(Whh[c], whhB[c], GDIM * HDIM);
        k_add_bias<<<GDIM / 256, 256, 0, stream>>>(bih[c], bhh[c], biasC[c], GDIM);
    }
    k_cvt_bf16<<<256, 256, 0, stream>>>(gwW,  gwWB,  HDIM * HDIM);
    k_cvt_bf16<<<512, 256, 0, stream>>>(mirW, mirWB, HDIM * 2 * HDIM);
    k_cvt_bf16<<<256, 256, 0, stream>>>(outW, outWB, OUT_F * HDIM);
    k_zero_f32<<<256, 256, 0, stream>>>(states, (int)(10 * BH));

    const dim3 blk(256);
    const dim3 gGate(GDIM / 128);   // 16 WGs: gate GEMMs, N=2048
    const dim3 gLin(HDIM / 128);    // 4 WGs:  linears,    N=512

    for (int t = 0; t < SEQ; ++t) {
        const float* xt = inputs + (size_t)t * BATCH * IN_F;
        float* yt = outp + (size_t)t * BATCH * OUT_F;

        // perc cell: x = input[t], h = h_perc
        k_gemm_dual<<<gGate, blk, 0, stream>>>(xt, IN_F, h_perc, HDIM,
            wihB[0], IN_F, whhB[0], HDIM, biasC[0], IN_F, HDIM, gates, GDIM, 0);
        k_nf_update<<<BATCH, blk, 0, stream>>>(gates, xt, IN_F,
            h_perc, c_perc, Av[0], wt[0], gm[0], bt[0]);

        // gw cell: x = perc, h = h_gw
        k_gemm_dual<<<gGate, blk, 0, stream>>>(h_perc, HDIM, h_gw, HDIM,
            wihB[1], HDIM, whhB[1], HDIM, biasC[1], HDIM, HDIM, gates, GDIM, 0);
        k_nf_update<<<BATCH, blk, 0, stream>>>(gates, h_perc, HDIM,
            h_gw, c_gw, Av[1], wt[1], gm[1], bt[1]);

        // decision = gw_out @ gw_out_W^T + b
        k_gemm_dual<<<gLin, blk, 0, stream>>>(h_gw, HDIM, nullptr, 0,
            gwWB, HDIM, nullptr, 0, gwb, HDIM, 0, dec, HDIM, 0);

        // obs cell: x = perc, h = h_obs
        k_gemm_dual<<<gGate, blk, 0, stream>>>(h_perc, HDIM, h_obs, HDIM,
            wihB[2], HDIM, whhB[2], HDIM, biasC[2], HDIM, HDIM, gates, GDIM, 0);
        k_nf_update<<<BATCH, blk, 0, stream>>>(gates, h_perc, HDIM,
            h_obs, c_obs, Av[2], wt[2], gm[2], bt[2]);

        // act cell (mir_act): x = decision, h = h_act
        k_gemm_dual<<<gGate, blk, 0, stream>>>(dec, HDIM, h_act, HDIM,
            wihB[3], HDIM, whhB[3], HDIM, biasC[3], HDIM, HDIM, gates, GDIM, 0);
        k_nf_update<<<BATCH, blk, 0, stream>>>(gates, dec, HDIM,
            h_act, c_act, Av[3], wt[3], gm[3], bt[3]);

        // mirror = sigmoid([obs, act] @ mirror_W^T + b)  (dual-K, fused sigmoid)
        k_gemm_dual<<<gLin, blk, 0, stream>>>(h_obs, HDIM, h_act, HDIM,
            mirWB, 2 * HDIM, mirWB + HDIM, 2 * HDIM, mirb, HDIM, HDIM, mir, HDIM, 1);

        // ae cell: x = mirror, h = h_ae
        k_gemm_dual<<<gGate, blk, 0, stream>>>(mir, HDIM, h_ae, HDIM,
            wihB[4], HDIM, whhB[4], HDIM, biasC[4], HDIM, HDIM, gates, GDIM, 0);
        k_nf_update<<<BATCH, blk, 0, stream>>>(gates, mir, HDIM,
            h_ae, c_ae, Av[4], wt[4], gm[4], bt[4]);

        // y = ae @ out_W^T + b  -> outputs[t]
        k_gemm_dual<<<gLin, blk, 0, stream>>>(h_ae, HDIM, nullptr, 0,
            outWB, HDIM, nullptr, 0, outb, HDIM, 0, yt, OUT_F, 0);
    }
}
